// BatchInfoNCELoss_8641474200145
// MI455X (gfx1250) — matrix operand order repaired and checked by reading, not compiled, not run on hardware
//
#include <hip/hip_runtime.h>
#include <hip/hip_bf16.h>

typedef __attribute__((ext_vector_type(16))) _Float16 v16h;
typedef __attribute__((ext_vector_type(8)))  _Float16 v8h;
typedef __attribute__((ext_vector_type(8)))  float    v8f;

#define BB   16
#define HH   128
#define WW   128
#define CC   16
#define PDIM 16384      // H*W
#define DD   144        // 3*3*16
#define DP   160        // D padded to 5 chunks of 32
#define SS   100
#define SP   112        // S padded to 7 tiles of 16
#define KK   8

__constant__ int c_offy[13] = {-2,-1,-1,-1, 0, 0, 0, 0, 0, 1, 1, 1, 2};
__constant__ int c_offx[13] = { 0,-1, 0, 1,-2,-1, 0, 1, 2,-1, 0, 1, 0};

// ---------------- zero the accumulator region ----------------
__global__ void zero_kernel(float* __restrict__ acc, int n) {
    int i = blockIdx.x * 256 + threadIdx.x;
    if (i < n) acc[i] = 0.0f;
}

// ---------------- build normalized f16 patches (edge-padded 3x3) ----------------
__global__ __launch_bounds__(128)
void patch_kernel(const float* __restrict__ latents, _Float16* __restrict__ patchesH) {
    int gid = blockIdx.x * 128 + threadIdx.x;       // b*P + p  (exact grid)
    int b = gid >> 14;
    int p = gid & (PDIM - 1);
    int py = p >> 7, px = p & (WW - 1);
    const float* base = latents + (size_t)b * HH * WW * CC;

    float norm2 = 0.0f;
    for (int dy = -1; dy <= 1; ++dy) {
        int yy = min(max(py + dy, 0), HH - 1);
        for (int dx = -1; dx <= 1; ++dx) {
            int xx = min(max(px + dx, 0), WW - 1);
            const float* src = base + ((size_t)yy * WW + xx) * CC;
            #pragma unroll
            for (int c = 0; c < CC; ++c) { float v = src[c]; norm2 += v * v; }
        }
    }
    float inv = 1.0f / fmaxf(sqrtf(norm2), 1e-12f);

    _Float16* dst = patchesH + (size_t)gid * DP;
    int i = 0;
    for (int dy = -1; dy <= 1; ++dy) {
        int yy = min(max(py + dy, 0), HH - 1);
        for (int dx = -1; dx <= 1; ++dx) {
            int xx = min(max(px + dx, 0), WW - 1);
            const float* src = base + ((size_t)yy * WW + xx) * CC;
            #pragma unroll
            for (int c = 0; c < CC; ++c) dst[i++] = (_Float16)(src[c] * inv);
        }
    }
    #pragma unroll
    for (int j = DD; j < DP; ++j) dst[j] = (_Float16)0.0f;
}

// ---------------- WMMA GEMM: sims = anchors x all patches, fused exp+mask accum ----------------
// grid: B * 256 blocks; block = 128 threads (4 waves); each wave: one 16-pixel N-tile,
// loops 7 anchor M-tiles, K = 5 chunks of 32 (D padded to 160 with zeros).
// Epilogue: shuffle-reduce across the 16 pixel lanes of each half-wave (anchor is
// uniform there), then a single LDS atomic per (m,v) from lane 0 / lane 16.
__global__ __launch_bounds__(128)
void gemm_within_kernel(const _Float16* __restrict__ patchesH,
                        const int* __restrict__ anchor_idx,
                        float* __restrict__ acc) {
    __shared__ float sPosS[SP], sPosC[SP], sNegS[SP], sNegC[SP];
    __shared__ int   sAidx[SP], sAy[SP], sAx[SP];

    int b    = blockIdx.x >> 8;          // 256 pixel-blocks per batch
    int pblk = blockIdx.x & 255;
    int t    = threadIdx.x;

    if (t < SP) {
        sPosS[t] = 0.0f; sPosC[t] = 0.0f; sNegS[t] = 0.0f; sNegC[t] = 0.0f;
        int idx = (t < SS) ? anchor_idx[b * SS + t] : -1;
        sAidx[t] = idx;
        sAy[t]   = idx >> 7;
        sAx[t]   = idx & (WW - 1);
    }
    __syncthreads();

    int wave  = t >> 5;
    int lane  = t & 31;
    int n     = lane & 15;
    int hi    = (lane >> 4) & 1;
    int pbase = (pblk * 4 + wave) * 16;
    int p     = pbase + n;

    // B operand: column N = lane%16 holds K contiguous (0..15 | 16..31) per chunk
    const _Float16* prow = patchesH + ((size_t)b * PDIM + p) * DP;
    v16h breg[5];
    #pragma unroll
    for (int kk = 0; kk < 5; ++kk)
        breg[kk] = *(const v16h*)(prow + kk * 32 + hi * 16);   // 32B aligned

    int py = p >> 7, px = p & (WW - 1);

    for (int m = 0; m < 7; ++m) {
        int sA   = m * 16 + n;           // A row M = lane%16
        int aidx = sAidx[sA];
        const _Float16* arow = patchesH + ((size_t)b * PDIM + max(aidx, 0)) * DP;

        v8f c = {};
        #pragma unroll
        for (int kk = 0; kk < 5; ++kk) {
            v16h a = {};
            if (aidx >= 0) {
                // A layout: halves 0..15 -> K = kbase + 16*hi + i (contiguous, matches B)
                a = *(const v16h*)(arow + kk * 32 + hi * 16);
            }
            c = __builtin_amdgcn_wmma_f32_16x16x32_f16(false, a, false, breg[kk],
                                                       (short)0, c, false, false);
        }

        // c[v] = sims[anchor = m*16 + v + 8*hi][pixel = p]; anchor uniform per half-wave
        #pragma unroll
        for (int v = 0; v < 8; ++v) {
            int s = m * 16 + v + hi * 8;
            if (s < SS) {                               // uniform within each half-wave
                int dy = py - sAy[s], dx = px - sAx[s];
                int d2 = dy * dy + dx * dx;
                bool pos = (d2 > 0) && (d2 <= 9);       // dist in (0, 3]
                bool neg = (d2 > 121);                  // dist > 11
                float e  = (pos || neg) ? __expf(c[v]) : 0.0f;   // T_WITHIN = 1
                float ep = pos ? e : 0.0f, cp = pos ? 1.0f : 0.0f;
                float en = neg ? e : 0.0f, cn = neg ? 1.0f : 0.0f;
                #pragma unroll
                for (int msk = 1; msk <= 8; msk <<= 1) {      // stays inside 16-lane half
                    ep += __shfl_xor(ep, msk, 32);
                    cp += __shfl_xor(cp, msk, 32);
                    en += __shfl_xor(en, msk, 32);
                    cn += __shfl_xor(cn, msk, 32);
                }
                if (n == 0) {                           // one lane per half-wave
                    if (cp > 0.0f) { atomicAdd(&sPosS[s], ep); atomicAdd(&sPosC[s], cp); }
                    if (cn > 0.0f) { atomicAdd(&sNegS[s], en); atomicAdd(&sNegC[s], cn); }
                }
            }
        }
    }
    __syncthreads();

    if (t < SP) {
        float* posS = acc;
        float* posC = acc + BB * SP;
        float* negS = acc + 2 * BB * SP;
        float* negC = acc + 3 * BB * SP;
        int j = b * SP + t;
        atomicAdd(&posS[j], sPosS[t]);
        atomicAdd(&posC[j], sPosC[t]);
        atomicAdd(&negS[j], sNegS[t]);
        atomicAdd(&negC[j], sNegC[t]);
    }
}

// ---------------- cross-image term: 8 images x 13 offsets per anchor ----------------
__global__ __launch_bounds__(128)
void cross_kernel(const _Float16* __restrict__ patchesH,
                  const int* __restrict__ anchor_idx,
                  const int* __restrict__ cross_idx,
                  float* __restrict__ acc) {
    __shared__ float sAnch[DD];
    __shared__ float red[128], redc[128];

    int b = blockIdx.x / SS;
    int s = blockIdx.x % SS;
    int t = threadIdx.x;

    int aidx = anchor_idx[b * SS + s];
    for (int d = t; d < DD; d += 128)
        sAnch[d] = (float)patchesH[((size_t)b * PDIM + aidx) * DP + d];
    __syncthreads();

    int ay = aidx >> 7, ax = aidx & (WW - 1);
    float esum = 0.0f, cnt = 0.0f;
    if (t < KK * 13) {
        int k = t / 13, m = t % 13;
        int pyy = ay + c_offy[m], pxx = ax + c_offx[m];
        bool valid = (pyy >= 0) && (pyy < HH) && (pxx >= 0) && (pxx < WW);
        int cy = min(max(pyy, 0), HH - 1);
        int cx = min(max(pxx, 0), WW - 1);
        int pidx = cy * WW + cx;
        int cb = cross_idx[(b * SS + s) * KK + k];
        const _Float16* row = patchesH + ((size_t)cb * PDIM + pidx) * DP;
        float dot = 0.0f;
        for (int d = 0; d < DD; ++d) dot += sAnch[d] * (float)row[d];
        if (valid) { esum = __expf(dot * 2.0f); cnt = 1.0f; }   // /T_ACROSS = *2
    }
    red[t] = esum; redc[t] = cnt;
    __syncthreads();
    for (int off = 64; off > 0; off >>= 1) {
        if (t < off) { red[t] += red[t + off]; redc[t] += redc[t + off]; }
        __syncthreads();
    }
    if (t == 0) {
        acc[4 * BB * SP + b * SP + s] = red[0];
        acc[5 * BB * SP + b * SP + s] = redc[0];
    }
}

// ---------------- finalize: per-anchor losses -> scalar mean ----------------
__global__ void final_kernel(const float* __restrict__ acc, float* __restrict__ out) {
    __shared__ float rsum[256], rcnt[256];
    int t = threadIdx.x;
    const float* posS = acc;
    const float* posC = acc + BB * SP;
    const float* negS = acc + 2 * BB * SP;
    const float* negC = acc + 3 * BB * SP;
    const float* crsS = acc + 4 * BB * SP;
    const float* crsC = acc + 5 * BB * SP;

    float lsum = 0.0f, lcnt = 0.0f;
    for (int i = t; i < BB * SS; i += 256) {
        int b = i / SS, s = i % SS;
        int j = b * SP + s;
        float pc = posC[j], nc = negC[j], cc = crsC[j];
        float pm = posS[j] / fmaxf(pc, 1.0f);
        float nm = negS[j] / fmaxf(nc, 1.0f);
        float cm = crsS[j] / fmaxf(cc, 1.0f);
        bool hasNeg = nc > 0.0f, hasCross = cc > 0.0f;
        if ((pc > 0.0f) && (hasNeg || hasCross)) {
            float per = 0.0f;
            if (hasNeg)   per += -logf(pm / (pm + nm + 1e-8f));
            if (hasCross) per += -logf(pm / (pm + cm + 1e-8f));
            lsum += per; lcnt += 1.0f;
        }
    }
    rsum[t] = lsum; rcnt[t] = lcnt;
    __syncthreads();
    for (int off = 128; off > 0; off >>= 1) {
        if (t < off) { rsum[t] += rsum[t + off]; rcnt[t] += rcnt[t + off]; }
        __syncthreads();
    }
    if (t == 0) out[0] = (rcnt[0] > 0.0f) ? (rsum[0] / rcnt[0]) : 0.0f;
}

extern "C" void kernel_launch(void* const* d_in, const int* in_sizes, int n_in,
                              void* d_out, int out_size, void* d_ws, size_t ws_size,
                              hipStream_t stream) {
    const float* latents    = (const float*)d_in[0];   // (16,128,128,16) f32
    const int*   anchor_idx = (const int*)d_in[1];     // (16,100)
    const int*   cross_idx  = (const int*)d_in[2];     // (16,100,8)
    float*       out        = (float*)d_out;           // scalar

    _Float16* patchesH = (_Float16*)d_ws;                                   // B*P*DP halfs = 80 MB
    float*    acc      = (float*)((char*)d_ws + (size_t)BB * PDIM * DP * sizeof(_Float16));
    int nacc = 6 * BB * SP;

    zero_kernel<<<(nacc + 255) / 256, 256, 0, stream>>>(acc, nacc);
    patch_kernel<<<(BB * PDIM) / 128, 128, 0, stream>>>(latents, patchesH);
    gemm_within_kernel<<<BB * 256, 128, 0, stream>>>(patchesH, anchor_idx, acc);
    cross_kernel<<<BB * SS, 128, 0, stream>>>(patchesH, anchor_idx, cross_idx, acc);
    final_kernel<<<1, 256, 0, stream>>>(acc, out);
}
